// MultiHeadedAttention_549755814076
// MI455X (gfx1250) — matrix operand order repaired and checked by reading, not compiled, not run on hardware
//
#include <hip/hip_runtime.h>
#include <math.h>

typedef _Float16 v16h __attribute__((ext_vector_type(16)));
typedef _Float16 v4h  __attribute__((ext_vector_type(4)));
typedef float    v8f  __attribute__((ext_vector_type(8)));

#define HEADS    8
#define HIDDEN   256
#define HEAD_DIM 32
#define BATCH    4
#define SEQ      2048
#define BS       (BATCH * SEQ)   // 8192 total rows
#define LOG2E    1.44269504088896340736f

// ---------------------------------------------------------------------------
// WMMA helper: D = A(16x32 f16) * B(32x16 f16) + C(16x16 f32)
// ---------------------------------------------------------------------------
__device__ __forceinline__ v8f wmma16(v16h a, v16h b, v8f c) {
  return __builtin_amdgcn_wmma_f32_16x16x32_f16(false, a, false, b,
                                                (short)0, c, false, false);
}

// A-fragment loader (16x32, MxK). ISA layout (CDNA5 ISA, 16-bit A 16x32):
//   lanes 0-15 : M=lane,    j=0..7 -> K=0..7,   j=8..15 -> K=16..23
//   lanes16-31 : M=lane-16, j=0..7 -> K=8..15,  j=8..15 -> K=24..31
__device__ __forceinline__ v16h load_afrag_f16(const _Float16* __restrict__ p, int lane) {
  const int b0 = (lane & 16) ? 8 : 0;
  v16h r;
#pragma unroll
  for (int j = 0; j < 8; ++j) r[j] = p[b0 + j];
#pragma unroll
  for (int j = 0; j < 8; ++j) r[j + 8] = p[b0 + 16 + j];
  return r;
}

// B-fragment loader (32x16, KxN):
//   lanes 0-15: N=lane, j=0..7 -> K=0..7, j=8..15 -> K=16..23; lanes 16-31: N=lane-16, j=0..7 -> K=8..15, j=8..15 -> K=24..31.
__device__ __forceinline__ v16h load_bfrag_f16(const _Float16* __restrict__ p, int lane) {
  const int b0 = (lane & 16) ? 8 : 0;
  v16h r;
#pragma unroll
  for (int j = 0; j < 8; ++j) r[j] = p[b0 + j];
#pragma unroll
  for (int j = 0; j < 8; ++j) r[j + 8] = p[b0 + 16 + j];
  return r;
}

// ---------------------------------------------------------------------------
// Kernel 0: one-shot fp32 -> f16 streaming conversion (vectorized 4-wide).
// Hoists all v_cvt out of the GEMM inner loops and halves their VMEM bytes.
// ---------------------------------------------------------------------------
__global__ __launch_bounds__(256) void cvt_f16_kernel(
    const float* __restrict__ src, _Float16* __restrict__ dst, int n4) {
  const int i = blockIdx.x * 256 + threadIdx.x;
  if (i < n4) {
    const float4 f = ((const float4*)src)[i];
    v4h h;
    h[0] = (_Float16)f.x;
    h[1] = (_Float16)f.y;
    h[2] = (_Float16)f.z;
    h[3] = (_Float16)f.w;
    ((v4h*)dst)[i] = h;
  }
}

// ---------------------------------------------------------------------------
// Kernel 1: fused Q/K/V projection.  out = X @ W^T + bias (f16 in, f16 out).
// Q,K stored [b*H+h][s][d]; V stored transposed [b*H+h][d][s].
// ---------------------------------------------------------------------------
__global__ __launch_bounds__(128) void qkv_proj_kernel(
    const _Float16* __restrict__ qf, const _Float16* __restrict__ kf,
    const _Float16* __restrict__ vf,
    const _Float16* __restrict__ Wqf, const float* __restrict__ bq,
    const _Float16* __restrict__ Wkf, const float* __restrict__ bk,
    const _Float16* __restrict__ Wvf, const float* __restrict__ bv,
    _Float16* __restrict__ Qh, _Float16* __restrict__ Kh,
    _Float16* __restrict__ Vt) {
  const int lane = threadIdx.x & 31;
  const int wave = threadIdx.x >> 5;
  const int ln15 = lane & 15;
  const int tile = blockIdx.x * 4 + wave;

  const int TILES_PER_MAT = (BS / 16) * (HIDDEN / 16);  // 512*16 = 8192
  const int mat = tile / TILES_PER_MAT;
  const int rem = tile % TILES_PER_MAT;
  const int mt = rem / (HIDDEN / 16);
  const int nt = rem % (HIDDEN / 16);

  const _Float16* X = (mat == 0) ? qf : (mat == 1) ? kf : vf;
  const _Float16* W = (mat == 0) ? Wqf : (mat == 1) ? Wkf : Wvf;
  const float* bias = (mat == 0) ? bq : (mat == 1) ? bk : bv;

  const int m0 = mt * 16;
  const int n0 = nt * 16;
  const int arow = m0 + ln15;
  const int bcol = n0 + ln15;

  v8f acc = {};
#pragma unroll
  for (int kk = 0; kk < HIDDEN; kk += 32) {
    v16h a = load_afrag_f16(X + (size_t)arow * HIDDEN + kk, lane);
    v16h b = load_bfrag_f16(W + (size_t)bcol * HIDDEN + kk, lane);
    acc = wmma16(a, b, acc);
  }

  const float bval = bias[bcol];
  const int h = bcol >> 5;
  const int d = bcol & 31;
  const int mbase = m0 + ((lane & 16) ? 8 : 0);

  if (mat < 2) {
    _Float16* dst = (mat == 0) ? Qh : Kh;
#pragma unroll
    for (int r = 0; r < 8; ++r) {
      const int m = mbase + r;
      const int b_ = m >> 11;        // / SEQ
      const int s_ = m & (SEQ - 1);
      dst[(((size_t)b_ * HEADS + h) * SEQ + s_) * HEAD_DIM + d] =
          (_Float16)(acc[r] + bval);
    }
  } else {
#pragma unroll
    for (int r = 0; r < 8; ++r) {
      const int m = mbase + r;
      const int b_ = m >> 11;
      const int s_ = m & (SEQ - 1);
      Vt[(((size_t)b_ * HEADS + h) * HEAD_DIM + d) * SEQ + s_] =
          (_Float16)(acc[r] + bval);
    }
  }
}

// ---------------------------------------------------------------------------
// Kernel 2: flash attention, transposed-score formulation (no LDS).
// One wave per (b,h,16-query tile). Scores computed as S^T = K * Q^T so the
// D-layout gives each lane ONE query (N=ln15) and 8 keys (M=vgpr): the P
// fragment for P*V is lane-local, and softmax stats need only a single
// shfl_xor(16) to combine key-halves.
// ---------------------------------------------------------------------------
__global__ __launch_bounds__(128) void attn_kernel(
    const _Float16* __restrict__ Qh, const _Float16* __restrict__ Kh,
    const _Float16* __restrict__ Vt, const int* __restrict__ mask,
    _Float16* __restrict__ hidden) {
  const int lane = threadIdx.x & 31;
  const int wave = threadIdx.x >> 5;
  const int ln15 = lane & 15;
  const int tile = blockIdx.x * 4 + wave;
  const int bh = tile >> 7;              // b*HEADS+h  (SEQ/16 = 128 q-tiles)
  const int qt = tile & 127;
  const int b_ = bh >> 3;
  const int h_ = bh & 7;
  const int q0 = qt * 16;
  const int mbase = (lane & 16) ? 8 : 0;

  // this lane's query row; mask==0 -> all-zero scores (uniform softmax,
  // identical result to the reference's -1e9 fill)
  const float keep = (mask[b_ * SEQ + q0 + ln15] != 0) ? 1.0f : 0.0f;

  // Q as B-fragment (N = query = ln15, k = d, d = b0..b0+7 and b0+16..b0+23)
  const v16h qfrag =
      load_bfrag_f16(Qh + ((size_t)bh * SEQ + q0 + ln15) * HEAD_DIM, lane);

  float mrun = -INFINITY, lrun = 0.0f;
  v8f o0 = {}, o1 = {};                  // O tile: N=d (lane), M=query (vgpr)
  const float scale = 0.17677669529663687f;  // 1/sqrt(32)

  for (int kx = 0; kx < SEQ; kx += 32) {
    // ---- S^T tiles: A = K rows (M=key, k=d), B = Q
    const _Float16* krow = Kh + ((size_t)bh * SEQ + kx + ln15) * HEAD_DIM;
    v16h ka0 = load_afrag_f16(krow, lane);                // keys kx   .. kx+15
    v16h ka1 = load_afrag_f16(krow + 16 * HEAD_DIM, lane);// keys kx+16.. kx+31
    v8f st0 = {}, st1 = {};
    st0 = wmma16(ka0, qfrag, st0);
    st1 = wmma16(ka1, qfrag, st1);

    // ---- lane-local masked/scaled scores for this lane's query
    float sv0[8], sv1[8];
    float lm = -INFINITY;
#pragma unroll
    for (int r = 0; r < 8; ++r) {
      sv0[r] = keep * (st0[r] * scale);   // key = kx + mbase + r
      sv1[r] = keep * (st1[r] * scale);   // key = kx + 16 + mbase + r
      lm = fmaxf(lm, fmaxf(sv0[r], sv1[r]));
    }
    lm = fmaxf(lm, __shfl_xor(lm, 16, 32));   // combine the two key-halves
    const float mnew = fmaxf(mrun, lm);

    // ---- P fragment is lane-local: A-layout keys b0..b0+7 / b0+16..b0+23
    // with b0 = mbase, exactly the keys this lane holds in st0/st1.
    v16h pfrag;
    float ls = 0.0f;
#pragma unroll
    for (int r = 0; r < 8; ++r) {
      const float p0 = exp2f((sv0[r] - mnew) * LOG2E);
      const float p1 = exp2f((sv1[r] - mnew) * LOG2E);
      ls += p0 + p1;
      pfrag[r]     = (_Float16)p0;
      pfrag[r + 8] = (_Float16)p1;
    }
    ls += __shfl_xor(ls, 16, 32);
    const float corr = exp2f((mrun - mnew) * LOG2E);
    lrun = lrun * corr + ls;
    mrun = mnew;

    // ---- rescale O: corr is per-query (lives in lane == query index);
    // O rows are queries mbase..mbase+7 -> 8 broadcasts
#pragma unroll
    for (int r = 0; r < 8; ++r) {
      const float cr = __shfl(corr, mbase + r, 32);
      o0[r] *= cr;
      o1[r] *= cr;
    }

    // ---- O += P * V  (B = Vt columns d, contiguous over keys)
    const _Float16* vcol = Vt + ((size_t)bh * HEAD_DIM + ln15) * SEQ + kx;
    v16h bv0 = load_bfrag_f16(vcol, lane);             // d = 0..15
    v16h bv1 = load_bfrag_f16(vcol + 16 * SEQ, lane);  // d = 16..31
    o0 = wmma16(pfrag, bv0, o0);
    o1 = wmma16(pfrag, bv1, o1);
  }

  // ---- normalize (1/l per query, broadcast into row layout) and store
  const float linv = 1.0f / lrun;
#pragma unroll
  for (int r = 0; r < 8; ++r) {
    const float li = __shfl(linv, mbase + r, 32);
    const size_t row = (size_t)(b_ * SEQ + q0 + mbase + r);
    hidden[row * HIDDEN + h_ * HEAD_DIM + ln15]      = (_Float16)(o0[r] * li);
    hidden[row * HIDDEN + h_ * HEAD_DIM + 16 + ln15] = (_Float16)(o1[r] * li);
  }
}

// ---------------------------------------------------------------------------
// Kernel 3: output projection. out = hidden(f16) @ Wo^T + bo, fp32 out.
// ---------------------------------------------------------------------------
__global__ __launch_bounds__(128) void out_proj_kernel(
    const _Float16* __restrict__ hidden, const _Float16* __restrict__ Wof,
    const float* __restrict__ bo, float* __restrict__ out) {
  const int lane = threadIdx.x & 31;
  const int wave = threadIdx.x >> 5;
  const int ln15 = lane & 15;
  const int tile = blockIdx.x * 4 + wave;
  const int mt = tile / (HIDDEN / 16);
  const int nt = tile % (HIDDEN / 16);
  const int arow = mt * 16 + ln15;
  const int bcol = nt * 16 + ln15;

  v8f acc = {};
#pragma unroll
  for (int kk = 0; kk < HIDDEN; kk += 32) {
    v16h a = load_afrag_f16(hidden + (size_t)arow * HIDDEN + kk, lane);
    v16h b = load_bfrag_f16(Wof + (size_t)bcol * HIDDEN + kk, lane);
    acc = wmma16(a, b, acc);
  }

  const float bval = bo[bcol];
  const int mbase = mt * 16 + ((lane & 16) ? 8 : 0);
#pragma unroll
  for (int r = 0; r < 8; ++r)
    out[(size_t)(mbase + r) * HIDDEN + bcol] = acc[r] + bval;
}

// ---------------------------------------------------------------------------
extern "C" void kernel_launch(void* const* d_in, const int* in_sizes, int n_in,
                              void* d_out, int out_size, void* d_ws,
                              size_t ws_size, hipStream_t stream) {
  (void)in_sizes; (void)n_in; (void)out_size; (void)ws_size;
  const float* q    = (const float*)d_in[0];
  const float* k    = (const float*)d_in[1];
  const float* v    = (const float*)d_in[2];
  const int*   mask = (const int*)d_in[3];
  const float* Wq   = (const float*)d_in[4];
  const float* bq   = (const float*)d_in[5];
  const float* Wk   = (const float*)d_in[6];
  const float* bk   = (const float*)d_in[7];
  const float* Wv   = (const float*)d_in[8];
  const float* bv   = (const float*)d_in[9];
  const float* Wo   = (const float*)d_in[10];
  const float* bo   = (const float*)d_in[11];
  float* out = (float*)d_out;

  // ---- workspace layout (all f16), total ~28.5 MB
  const size_t n_x = (size_t)BS * HIDDEN;                       // 2,097,152
  const size_t n_w = (size_t)HIDDEN * HIDDEN;                   //    65,536
  const size_t per_mat = (size_t)BATCH * HEADS * SEQ * HEAD_DIM;// 2,097,152
  _Float16* qf  = (_Float16*)d_ws;
  _Float16* kf  = qf + n_x;
  _Float16* vf  = kf + n_x;
  _Float16* Wqf = vf + n_x;
  _Float16* Wkf = Wqf + n_w;
  _Float16* Wvf = Wkf + n_w;
  _Float16* Wof = Wvf + n_w;
  _Float16* Qh  = Wof + n_w;
  _Float16* Kh  = Qh + per_mat;
  _Float16* Vt  = Kh + per_mat;
  _Float16* hid = Vt + per_mat;

  // ---- one-shot f32 -> f16 conversions (streaming, 4-wide)
  const int nx4 = (int)(n_x / 4);   // 524,288 -> 2048 blocks
  const int nw4 = (int)(n_w / 4);   //  16,384 ->   64 blocks
  cvt_f16_kernel<<<nx4 / 256, 256, 0, stream>>>(q, qf, nx4);
  cvt_f16_kernel<<<nx4 / 256, 256, 0, stream>>>(k, kf, nx4);
  cvt_f16_kernel<<<nx4 / 256, 256, 0, stream>>>(v, vf, nx4);
  cvt_f16_kernel<<<nw4 / 256, 256, 0, stream>>>(Wq, Wqf, nw4);
  cvt_f16_kernel<<<nw4 / 256, 256, 0, stream>>>(Wk, Wkf, nw4);
  cvt_f16_kernel<<<nw4 / 256, 256, 0, stream>>>(Wv, Wvf, nw4);
  cvt_f16_kernel<<<nw4 / 256, 256, 0, stream>>>(Wo, Wof, nw4);

  // 3 mats * 512 * 16 tiles / 4 waves = 6144 blocks
  qkv_proj_kernel<<<6144, 128, 0, stream>>>(qf, kf, vf, Wqf, bq, Wkf, bk,
                                            Wvf, bv, Qh, Kh, Vt);
  // 4*8*128 q-tiles / 4 waves = 1024 blocks
  attn_kernel<<<1024, 128, 0, stream>>>(Qh, Kh, Vt, mask, hid);
  // 512 * 16 tiles / 4 waves = 2048 blocks
  out_proj_kernel<<<2048, 128, 0, stream>>>(hid, Wof, bo, out);
}
